// EncoderRNN_75548474736872
// MI455X (gfx1250) — matrix operand order repaired and checked by reading, not compiled, not run on hardware
//
#include <hip/hip_runtime.h>
#include <hip/hip_bf16.h>

// ---------------------------------------------------------------------------
// EncoderRNN for MI455X (gfx1250, wave32)
//   Stage 1: char-GRU, batch=2048, 24 steps, fused [x|h] GEMM via
//            v_wmma_f32_16x16x32_bf16; B tiles via global_load_async_to_lds
//            (K loop split into x-phase / h-phase to avoid accumulator phis)
//   Stage 2: input GEMMs gi2 = word_emb @ Wih2^T (+bias) via WMMA + async LDS
//   Stage 3: persistent dual-direction word-GRU recurrence (batch=1,
//            2048 steps) with per-group atomic grid barriers
// ---------------------------------------------------------------------------

typedef __bf16 bf16_t;
typedef bf16_t v16bf __attribute__((ext_vector_type(16)));
typedef float  v8f   __attribute__((ext_vector_type(8)));

#define W_WORDS 2048
#define CCH     24
#define EDIM    512
#define H1      1024
#define H2      1024
#define G2BLK   32      // blocks per direction in the persistent kernel

__device__ __forceinline__ bf16_t f2bf(float f) {
    unsigned u = __float_as_uint(f);
    unsigned r = u + 0x7FFFu + ((u >> 16) & 1u);   // round-to-nearest-even
    unsigned short h = (unsigned short)(r >> 16);
    return __builtin_bit_cast(bf16_t, h);
}

__device__ __forceinline__ float sigmoidf_(float x) {
    return 1.0f / (1.0f + __expf(-x));
}

// A-tile column mapping: store K columns contiguously (k = 16h + s), matching
// the contiguous B fragment order; each lane's 16 halves are contiguous 32B.
__device__ __forceinline__ int aperm(int k) {
    int g  = k >> 3;                    // 0..3
    int ng = ((g & 1) << 1) | (g >> 1); // 0,2,1,3
    (void)ng; return k;
}

// Async global->LDS copy of one 16-byte chunk (CDNA5, tracked by ASYNCcnt).
__device__ __forceinline__ void async_b128(unsigned lds_off, const void* gsrc) {
    asm volatile("global_load_async_to_lds_b128 %0, %1, off"
                 :: "v"(lds_off), "v"((unsigned long long)(size_t)gsrc)
                 : "memory");
}
__device__ __forceinline__ void wait_async0() {
    asm volatile("s_wait_asynccnt 0x0" ::: "memory");
}

#define WMMA_BF16(ACC, AA, BB) \
    (ACC) = __builtin_amdgcn_wmma_f32_16x16x32_bf16(false, (AA), false, (BB), (short)0, (ACC), false, false)

// ---------------------------------------------------------------------------
// Stage 1: one GRU time step, fused GEMM + gates.
// Grid: (H1/32, W/128), block 256 threads = 8 waves.
// Each block: 128 words x 32 hidden units; K = 512 (x via emb gather) + 1024 (h).
// ---------------------------------------------------------------------------
__global__ __launch_bounds__(256) void gru1_step(
    const int*    __restrict__ x,      // (W, C) char indices
    const float*  __restrict__ emb,    // (V, E)
    const bf16_t* __restrict__ Wih,    // (3H1, E)  bf16
    const bf16_t* __restrict__ Whh,    // (3H1, H1) bf16
    const float*  __restrict__ bih,
    const float*  __restrict__ bhh,
    const float*  __restrict__ hprev,  // (W, H1)
    float*        __restrict__ hnext,  // (W, H1)
    float*        __restrict__ word_emb_out, // d_out on last step, else null
    int t)
{
    __shared__ __align__(64) bf16_t As[128][48];  // [m][perm(k)], 96B rows
    __shared__ __align__(64) bf16_t Bs[96][32];   // [gate*32 + n][k], 64B rows

    const int tid  = threadIdx.x;
    const int lane = tid & 31;
    const int wv   = tid >> 5;
    const int n0   = blockIdx.x * 32;   // hidden-unit base
    const int m0   = blockIdx.y * 128;  // word base

    v8f zero = {0.f,0.f,0.f,0.f,0.f,0.f,0.f,0.f};
    v8f accr[2], accz[2], accx[2], acch[2];
    accr[0]=zero; accr[1]=zero; accz[0]=zero; accz[1]=zero;
    accx[0]=zero; accx[1]=zero; acch[0]=zero; acch[1]=zero;

    const int ar  = wv * 16 + (lane & 15);
    const int akb = (lane >> 4) * 16;
    const int bn  = lane & 15;
    const int bkb = (lane >> 4) * 16;

    #define LDB(ROWBASE) (*(const v16bf*)&Bs[(ROWBASE) + bn][bkb])

    // ---- Phase 1: K = 0..511 (x inputs via embedding gather, Wih) ---------
    for (int kk = 0; kk < 16; ++kk) {
        const int kg = kk * 32;
        // async-stage B tile: 96 rows x 32 halfs = 384 x 16B chunks
        {
            int row = tid >> 2, part = tid & 3;
            int grow = (row >> 5) * H1 + n0 + (row & 31);
            async_b128((unsigned)(size_t)&Bs[row][part * 8],
                       Wih + (size_t)grow * EDIM + kg + part * 8);
        }
        if (tid < 128) {
            int c2 = 256 + tid;
            int row = c2 >> 2, part = c2 & 3;
            int grow = (row >> 5) * H1 + n0 + (row & 31);
            async_b128((unsigned)(size_t)&Bs[row][part * 8],
                       Wih + (size_t)grow * EDIM + kg + part * 8);
        }
        // VALU-stage A tile (embedding gather, fp32 -> bf16, k contiguous)
        for (int i = tid; i < 128 * 32; i += 256) {
            int m = i >> 5, k = i & 31;
            int idx = x[(m0 + m) * CCH + t];
            As[m][aperm(k)] = f2bf(emb[(size_t)idx * EDIM + kg + k]);
        }
        wait_async0();
        __syncthreads();

        v16bf a = *(const v16bf*)&As[ar][akb];
        { v16bf b = LDB(0);  WMMA_BF16(accr[0], a, b); }
        { v16bf b = LDB(16); WMMA_BF16(accr[1], a, b); }
        { v16bf b = LDB(32); WMMA_BF16(accz[0], a, b); }
        { v16bf b = LDB(48); WMMA_BF16(accz[1], a, b); }
        { v16bf b = LDB(64); WMMA_BF16(accx[0], a, b); }
        { v16bf b = LDB(80); WMMA_BF16(accx[1], a, b); }
        __syncthreads();
    }

    // ---- Phase 2: K = 512..1535 (hidden state, Whh) -----------------------
    for (int kk = 0; kk < 32; ++kk) {
        const int kg = kk * 32;
        {
            int row = tid >> 2, part = tid & 3;
            int grow = (row >> 5) * H1 + n0 + (row & 31);
            async_b128((unsigned)(size_t)&Bs[row][part * 8],
                       Whh + (size_t)grow * H1 + kg + part * 8);
        }
        if (tid < 128) {
            int c2 = 256 + tid;
            int row = c2 >> 2, part = c2 & 3;
            int grow = (row >> 5) * H1 + n0 + (row & 31);
            async_b128((unsigned)(size_t)&Bs[row][part * 8],
                       Whh + (size_t)grow * H1 + kg + part * 8);
        }
        for (int i = tid; i < 128 * 32; i += 256) {
            int m = i >> 5, k = i & 31;
            As[m][aperm(k)] = f2bf(hprev[(size_t)(m0 + m) * H1 + kg + k]);
        }
        wait_async0();
        __syncthreads();

        v16bf a = *(const v16bf*)&As[ar][akb];
        { v16bf b = LDB(0);  WMMA_BF16(accr[0], a, b); }
        { v16bf b = LDB(16); WMMA_BF16(accr[1], a, b); }
        { v16bf b = LDB(32); WMMA_BF16(accz[0], a, b); }
        { v16bf b = LDB(48); WMMA_BF16(accz[1], a, b); }
        { v16bf b = LDB(64); WMMA_BF16(acch[0], a, b); }
        { v16bf b = LDB(80); WMMA_BF16(acch[1], a, b); }
        __syncthreads();
    }
    #undef LDB

    // Epilogue: gates + h update. C/D layout: VGPR i, lanes<16 -> M=i, N=lane;
    // lanes>=16 -> M=8+i, N=lane-16.
    const int mbase = m0 + wv * 16 + ((lane >> 4) * 8);
    const int nl    = lane & 15;
    #pragma unroll
    for (int s = 0; s < 2; ++s) {
        int j = n0 + s * 16 + nl;
        float br = bih[j] + bhh[j];
        float bz = bih[H1 + j] + bhh[H1 + j];
        float bx = bih[2 * H1 + j];
        float bh = bhh[2 * H1 + j];
        #pragma unroll
        for (int i = 0; i < 8; ++i) {
            int m = mbase + i;
            float r  = sigmoidf_(accr[s][i] + br);
            float z  = sigmoidf_(accz[s][i] + bz);
            float nn = tanhf((accx[s][i] + bx) + r * (acch[s][i] + bh));
            float hp = hprev[(size_t)m * H1 + j];
            float hv = (1.0f - z) * nn + z * hp;
            hnext[(size_t)m * H1 + j] = hv;
            if (word_emb_out) word_emb_out[(size_t)m * H1 + j] = hv;
        }
    }
}

// ---------------------------------------------------------------------------
// Stage 2: gi2 = word_emb @ Wih2^T + bih2   (2048x1024 @ 1024x3072)
// Grid: (3072/64, 2048/128), 256 threads, wave = 16-row strip x 64 cols.
// ---------------------------------------------------------------------------
__global__ __launch_bounds__(256) void gemm_in2(
    const float*  __restrict__ A,     // (2048, 1024) fp32 word_emb
    const bf16_t* __restrict__ Wb,    // (3072, 1024) bf16
    const float*  __restrict__ bias,  // (3072)
    float*        __restrict__ out)   // (2048, 3072)
{
    __shared__ __align__(64) bf16_t As[128][48];
    __shared__ __align__(64) bf16_t Bs[64][32];

    const int tid  = threadIdx.x;
    const int lane = tid & 31;
    const int wv   = tid >> 5;
    const int n0   = blockIdx.x * 64;
    const int m0   = blockIdx.y * 128;

    v8f zero = {0.f,0.f,0.f,0.f,0.f,0.f,0.f,0.f};
    v8f acc[4]; acc[0]=zero; acc[1]=zero; acc[2]=zero; acc[3]=zero;

    for (int kk = 0; kk < 32; ++kk) {
        const int kg = kk * 32;

        // async-stage B tile: 64 rows x 4 chunks = 256 = one chunk per thread
        {
            int row = tid >> 2, part = tid & 3;
            async_b128((unsigned)(size_t)&Bs[row][part * 8],
                       Wb + (size_t)(n0 + row) * H1 + kg + part * 8);
        }
        // VALU-stage A tile (fp32 -> bf16, k contiguous)
        for (int i = tid; i < 128 * 32; i += 256) {
            int m = i >> 5, k = i & 31;
            As[m][aperm(k)] = f2bf(A[(size_t)(m0 + m) * H1 + kg + k]);
        }
        wait_async0();
        __syncthreads();

        int ar = wv * 16 + (lane & 15);
        v16bf a = *(const v16bf*)&As[ar][(lane >> 4) * 16];

        int bn  = lane & 15;
        int bkb = (lane >> 4) * 16;
        #pragma unroll
        for (int s = 0; s < 4; ++s) {
            v16bf b = *(const v16bf*)&Bs[s * 16 + bn][bkb];
            WMMA_BF16(acc[s], a, b);
        }
        __syncthreads();
    }

    const int mbase = m0 + wv * 16 + ((lane >> 4) * 8);
    const int nl    = lane & 15;
    #pragma unroll
    for (int s = 0; s < 4; ++s) {
        int j = n0 + s * 16 + nl;
        float bj = bias[j];
        #pragma unroll
        for (int i = 0; i < 8; ++i)
            out[(size_t)(mbase + i) * (3 * H2) + j] = acc[s][i] + bj;
    }
}

// ---------------------------------------------------------------------------
// Stage 3: persistent word-GRU, fwd (blocks 0..31) + bwd (blocks 32..63)
// concurrently. Each group: 32 blocks x 256 threads; each block owns 32
// hidden units (96 gate rows); per step each wave does 12 dot products of
// length 1024 (float4 loads) with shuffle reduction; one grid barrier/step.
// ---------------------------------------------------------------------------
__global__ __launch_bounds__(256) void gru2_persistent(
    const float* __restrict__ gi_f,  const float* __restrict__ gi_b,   // (2048, 3072)
    const float* __restrict__ Whh_f, const float* __restrict__ Whh_b,  // (3072, 1024)
    const float* __restrict__ bhh_f, const float* __restrict__ bhh_b,
    float*       __restrict__ hbuf,  // 2 dirs * 2 bufs * H2
    unsigned*    __restrict__ bar,   // [dir*2 + {cnt,gen}]
    float*       __restrict__ ctx)   // (2048, 2*H2) = d_out + W*H1
{
    const int dir = blockIdx.x / G2BLK;
    const int gb  = blockIdx.x % G2BLK;
    const float* gi  = dir ? gi_b  : gi_f;
    const float* Whh = dir ? Whh_b : Whh_f;
    const float* bhh = dir ? bhh_b : bhh_f;
    float*    hbase = hbuf + dir * 2 * H2;
    unsigned* cnt   = bar + dir * 2;
    unsigned* gen   = bar + dir * 2 + 1;

    const int tid  = threadIdx.x;
    const int lane = tid & 31;
    const int wv   = tid >> 5;
    const int j0   = gb * 32;

    __shared__ __align__(16) float hs[H2];
    __shared__ float ghs[96];

    for (int s = 0; s < W_WORDS; ++s) {
        const int w = dir ? (W_WORDS - 1 - s) : s;
        const float* hc = hbase + (s & 1) * H2;
        float*       hn = hbase + ((s & 1) ^ 1) * H2;

        for (int i = tid; i < H2; i += 256) hs[i] = hc[i];
        __syncthreads();

        const float4* hs4 = (const float4*)hs;
        // 96 gate rows for this block; wave wv owns rows wv*12 .. wv*12+11
        for (int qq = 0; qq < 12; ++qq) {
            int q = wv * 12 + qq;
            int g = q >> 5;
            int j = j0 + (q & 31);
            const float4* wrow4 = (const float4*)(Whh + (size_t)(g * H2 + j) * H2);
            float acc = 0.f;
            for (int k = lane; k < H2 / 4; k += 32) {
                float4 wv4 = wrow4[k];
                float4 hv4 = hs4[k];
                acc += wv4.x * hv4.x + wv4.y * hv4.y + wv4.z * hv4.z + wv4.w * hv4.w;
            }
            #pragma unroll
            for (int m = 16; m; m >>= 1) acc += __shfl_xor(acc, m, 32);
            if (lane == 0) ghs[q] = acc + bhh[g * H2 + j];
        }
        __syncthreads();

        if (tid < 32) {
            int j = j0 + tid;
            const float* gir = gi + (size_t)w * (3 * H2);
            float r   = sigmoidf_(gir[j]      + ghs[tid]);
            float z   = sigmoidf_(gir[H2 + j] + ghs[32 + tid]);
            float nn  = tanhf(gir[2 * H2 + j] + r * ghs[64 + tid]);
            float hv  = (1.0f - z) * nn + z * hs[j];
            hn[j] = hv;
            ctx[(size_t)w * (2 * H2) + dir * H2 + j] = hv;
        }

        // per-group grid barrier (arrive + spin on generation)
        __syncthreads();
        if (tid == 0) {
            __threadfence();
            unsigned g0 = __hip_atomic_load(gen, __ATOMIC_ACQUIRE, __HIP_MEMORY_SCOPE_AGENT);
            unsigned v  = __hip_atomic_fetch_add(cnt, 1u, __ATOMIC_ACQ_REL, __HIP_MEMORY_SCOPE_AGENT);
            if (v == (unsigned)(G2BLK - 1)) {
                __hip_atomic_store(cnt, 0u, __ATOMIC_RELAXED, __HIP_MEMORY_SCOPE_AGENT);
                __hip_atomic_fetch_add(gen, 1u, __ATOMIC_RELEASE, __HIP_MEMORY_SCOPE_AGENT);
            } else {
                while (__hip_atomic_load(gen, __ATOMIC_ACQUIRE, __HIP_MEMORY_SCOPE_AGENT) == g0)
                    __builtin_amdgcn_s_sleep(1);
            }
        }
        __syncthreads();
    }
}

// ---------------------------------------------------------------------------
// Utility kernels
// ---------------------------------------------------------------------------
__global__ void cvt_bf16(const float* __restrict__ in, bf16_t* __restrict__ out, size_t n) {
    size_t i = (size_t)blockIdx.x * blockDim.x + threadIdx.x;
    size_t stride = (size_t)gridDim.x * blockDim.x;
    for (size_t k = i; k < n; k += stride) out[k] = f2bf(in[k]);
}

__global__ void init_ws(float* __restrict__ h1buf, size_t h1n,
                        float* __restrict__ h2buf, unsigned* __restrict__ bar) {
    size_t i = (size_t)blockIdx.x * blockDim.x + threadIdx.x;
    size_t stride = (size_t)gridDim.x * blockDim.x;
    for (size_t k = i; k < h1n; k += stride) h1buf[k] = 0.f;
    for (size_t k = i; k < 4 * (size_t)H2; k += stride) h2buf[k] = 0.f;
    if (i < 4) bar[i] = 0u;
}

// ---------------------------------------------------------------------------
extern "C" void kernel_launch(void* const* d_in, const int* in_sizes, int n_in,
                              void* d_out, int out_size, void* d_ws, size_t ws_size,
                              hipStream_t stream) {
    (void)in_sizes; (void)n_in; (void)out_size; (void)ws_size;

    const int*   x     = (const int*)  d_in[0];
    const float* emb   = (const float*)d_in[1];
    const float* Wih1  = (const float*)d_in[2];
    const float* Whh1  = (const float*)d_in[3];
    const float* bih1  = (const float*)d_in[4];
    const float* bhh1  = (const float*)d_in[5];
    const float* Wih2f = (const float*)d_in[6];
    const float* Whh2f = (const float*)d_in[7];
    const float* bih2f = (const float*)d_in[8];
    const float* bhh2f = (const float*)d_in[9];
    const float* Wih2b = (const float*)d_in[10];
    const float* Whh2b = (const float*)d_in[11];
    const float* bih2b = (const float*)d_in[12];
    const float* bhh2b = (const float*)d_in[13];
    float* out = (float*)d_out;

    // workspace carve-up
    char* ws = (char*)d_ws;
    size_t off = 0;
    auto carve = [&](size_t bytes) -> char* {
        char* p = ws + off;
        off = (off + bytes + 255) & ~(size_t)255;
        return p;
    };
    bf16_t* wih1_bf  = (bf16_t*)carve((size_t)3 * H1 * EDIM * 2);
    bf16_t* whh1_bf  = (bf16_t*)carve((size_t)3 * H1 * H1 * 2);
    bf16_t* wih2f_bf = (bf16_t*)carve((size_t)3 * H2 * H1 * 2);
    bf16_t* wih2b_bf = (bf16_t*)carve((size_t)3 * H2 * H1 * 2);
    float*  h1buf    = (float*) carve((size_t)2 * W_WORDS * H1 * 4);
    float*  gi2f     = (float*) carve((size_t)W_WORDS * 3 * H2 * 4);
    float*  gi2b     = (float*) carve((size_t)W_WORDS * 3 * H2 * 4);
    float*  h2buf    = (float*) carve((size_t)4 * H2 * 4);
    unsigned* bar    = (unsigned*)carve(64);

    const size_t NH1 = (size_t)W_WORDS * H1;

    init_ws<<<1024, 256, 0, stream>>>(h1buf, 2 * NH1, h2buf, bar);

    cvt_bf16<<<512, 256, 0, stream>>>(Wih1,  wih1_bf,  (size_t)3 * H1 * EDIM);
    cvt_bf16<<<512, 256, 0, stream>>>(Whh1,  whh1_bf,  (size_t)3 * H1 * H1);
    cvt_bf16<<<512, 256, 0, stream>>>(Wih2f, wih2f_bf, (size_t)3 * H2 * H1);
    cvt_bf16<<<512, 256, 0, stream>>>(Wih2b, wih2b_bf, (size_t)3 * H2 * H1);

    // Stage 1: 24 sequential GRU steps (ping-pong h buffers)
    for (int t = 0; t < CCH; ++t) {
        const float* hp = h1buf + (size_t)(t & 1) * NH1;
        float*       hq = h1buf + (size_t)((t + 1) & 1) * NH1;
        float* we_out = (t == CCH - 1) ? out : nullptr;  // word_emb -> d_out[0:W*H1]
        gru1_step<<<dim3(H1 / 32, W_WORDS / 128), 256, 0, stream>>>(
            x, emb, wih1_bf, whh1_bf, bih1, bhh1, hp, hq, we_out, t);
    }
    // final h (== word_emb) lives in h1buf buffer 0

    // Stage 2: input GEMMs for both directions
    gemm_in2<<<dim3(3 * H2 / 64, W_WORDS / 128), 256, 0, stream>>>(h1buf, wih2f_bf, bih2f, gi2f);
    gemm_in2<<<dim3(3 * H2 / 64, W_WORDS / 128), 256, 0, stream>>>(h1buf, wih2b_bf, bih2b, gi2b);

    // Stage 3: persistent dual-direction recurrence; ctx -> d_out[W*H1 : ]
    gru2_persistent<<<2 * G2BLK, 256, 0, stream>>>(
        gi2f, gi2b, Whh2f, Whh2b, bhh2f, bhh2b, h2buf, bar, out + NH1);
}
